// WaveNet_44727789420677
// MI455X (gfx1250) — matrix operand order repaired and checked by reading, not compiled, not run on hardware
//
#include <hip/hip_runtime.h>
#include <hip/hip_bf16.h>

typedef __attribute__((ext_vector_type(16))) _Float16 v16h;
typedef __attribute__((ext_vector_type(8)))  float    v8f;
typedef _Float16 f16;

#define B_      16
#define T0_     6139
#define CIN_    32
#define CR_     64
#define TPAD_   6160     // padded time length; max read index 6153 < 6160
#define SKIPN_  1024
#define DIMFC_  512
#define OUTN_   128
#define NBLK_   50
#define WBLKH_  16384    // halves per block: wd 8192 | wr 4096 | wk 4096

// ---------------------------------------------------------------------------
// WMMA fragment loaders (wave32, f16 16x16x32).
// A 16x32 (MxK): lane holds row M = lane&15; K pairs interleaved by lane-half:
//   lanes 0-15: V0..V3 -> K 0..7, V4..V7 -> K 16..23 ; lanes 16-31: +8.
// B 32x16 (KxN): lane holds col N = lane&15; same K interleave as A:
//   lanes 0-15: V0..V3 -> K 0..7, V4..V7 -> K 16..23 ; lanes 16-31: +8.
// row32 = pointer to this lane's 32-half, K-contiguous row segment.
// Both patterns are dword-contiguous per lane -> compiler emits b128 loads.
// ---------------------------------------------------------------------------
__device__ __forceinline__ v16h ld_fragA(const f16* row32, int lane) {
  union { v16h h; unsigned u[8]; } r;
  const unsigned* p = reinterpret_cast<const unsigned*>(row32);
  const int hi4 = (lane >> 4) << 2;
#pragma unroll
  for (int v = 0; v < 8; ++v)
    r.u[v] = p[(v & 3) + hi4 + ((v >> 2) << 3)];
  return r.h;
}

__device__ __forceinline__ v16h ld_fragB(const f16* row32, int lane) {
  union { v16h h; unsigned u[8]; } r;
  const unsigned* p = reinterpret_cast<const unsigned*>(row32);
  const int hi4 = (lane >> 4) << 2;
#pragma unroll
  for (int v = 0; v < 8; ++v)
    r.u[v] = p[(v & 3) + hi4 + ((v >> 2) << 3)];
  return r.h;
}

// gated activation: tanh(y)*sigmoid(y) == (1-u)/(1+u^2), u = exp(-y).
// Clamp y >= -30 (true value there ~ -1e-13) so u^2 stays finite in f32.
__device__ __forceinline__ float gated(float y) {
  const float u = __expf(-fmaxf(y, -30.0f));
  return (1.0f - u) * __builtin_amdgcn_rcpf(1.0f + u * u);
}

// ---------------------------------------------------------------------------
// One-time weight repack: f32 -> f16, per block [wd 64x128 | wr 64x64 | wk 64x64]
// wd[o][tap*64+ci] = Wdil[i][o][ci][tap]
// ---------------------------------------------------------------------------
__global__ __launch_bounds__(256)
void k_wcvt(const float* __restrict__ Wdil, const float* __restrict__ Wres,
            const float* __restrict__ Wskip, f16* __restrict__ wAll) {
  const int total = NBLK_ * WBLKH_;
  const int idx = blockIdx.x * 256 + threadIdx.x;
  if (idx >= total) return;
  const int i = idx >> 14;
  const int r = idx & (WBLKH_ - 1);
  float v;
  if (r < 8192) {
    const int o = r >> 7, kk = r & 127, tap = kk >> 6, ci = kk & 63;
    v = Wdil[(size_t)i * 8192 + (o * 64 + ci) * 2 + tap];
  } else if (r < 12288) {
    v = Wres[(size_t)i * 4096 + (r - 8192)];
  } else {
    v = Wskip[(size_t)i * 4096 + (r - 12288)];
  }
  wAll[idx] = (f16)v;
}

// ---------------------------------------------------------------------------
// Causal conv (k=2, left pad 1): x [B][T][32] f32 -> h [B][TPAD][64] f16
// ---------------------------------------------------------------------------
__global__ __launch_bounds__(256)
void k_causal(const float* __restrict__ x, const float* __restrict__ Wc,
              f16* __restrict__ h) {
  const int total = B_ * T0_ * CR_;
  int idx = blockIdx.x * 256 + threadIdx.x;
  if (idx >= total) return;
  const int c = idx & 63;
  const int t = (idx >> 6) % T0_;
  const int b = idx / (T0_ * CR_);
  const float* xc = x + ((size_t)b * T0_ + t) * CIN_;
  float a = 0.0f;
#pragma unroll
  for (int ci = 0; ci < CIN_; ++ci)
    a += Wc[(c * CIN_ + ci) * 2 + 1] * xc[ci];          // tap1 @ x[t]
  if (t > 0) {
    const float* xm = xc - CIN_;
#pragma unroll
    for (int ci = 0; ci < CIN_; ++ci)
      a += Wc[(c * CIN_ + ci) * 2 + 0] * xm[ci];        // tap0 @ x[t-1]
  }
  h[((size_t)b * TPAD_ + t) * 64 + c] = (f16)a;
}

// ---------------------------------------------------------------------------
// One WaveNet block: y = Wdil ⊛ h ; g = tanh(y)*sigmoid(y) ;
// hOut = Wres@g + bres + h[t+d] ; skip += (Wskip@g + bskip)[last 1024]
// grid: (ceil(Tout/128), B), block 256 (8 waves, 16 timesteps each).
// ---------------------------------------------------------------------------
__global__ __launch_bounds__(256)
void k_block(const f16* __restrict__ hIn, f16* __restrict__ hOut,
             float* __restrict__ skip, const f16* __restrict__ wblk,
             const float* __restrict__ bres, const float* __restrict__ bskip,
             int d, int Tout, int skipOff) {
  __shared__ f16 wlds[WBLKH_];      // [wd 64x128 | wr 64x64 | wk 64x64]
  __shared__ f16 gl[8 * 16 * 64];   // per-wave gated activations [n][c]

  const int tid = threadIdx.x;
  {
    const int4* src = reinterpret_cast<const int4*>(wblk);
    int4* dst = reinterpret_cast<int4*>(wlds);
#pragma unroll
    for (int i = 0; i < 8; ++i) dst[tid + i * 256] = src[tid + i * 256];
  }
  __syncthreads();
  const f16* wd = wlds;
  const f16* wr = wlds + 8192;
  const f16* wk = wlds + 12288;

  const int lane = tid & 31, wave = tid >> 5;
  const int t0 = (blockIdx.x * 8 + wave) * 16;
  const int b  = blockIdx.y;
  if (t0 >= Tout) return;                 // wave-uniform; no barriers after this

  const int n  = lane & 15;               // timestep within tile / A-row M
  const int hi = lane >> 4;

  // ---- stage 1: y[64 x 16] = Wcat[64 x 128] @ [h(t); h(t+d)] -------------
  v8f acc[4];
#pragma unroll
  for (int m = 0; m < 4; ++m)
#pragma unroll
    for (int r = 0; r < 8; ++r) acc[m][r] = 0.0f;

#pragma unroll
  for (int kf = 0; kf < 4; ++kf) {
    const int tap = kf >> 1, cb = (kf & 1) * 32;
    const f16* rowp =
        hIn + ((size_t)b * TPAD_ + (size_t)(t0 + n + tap * d)) * 64 + cb;
    const v16h bf = ld_fragB(rowp, lane);
#pragma unroll
    for (int m = 0; m < 4; ++m) {
      const v16h af = ld_fragA(&wd[(m * 16 + n) * 128 + kf * 32], lane);
      acc[m] = __builtin_amdgcn_wmma_f32_16x16x32_f16(
          false, af, false, bf, (short)0, acc[m], false, false);
    }
  }

  // ---- gated activation, pack f16 into per-wave LDS tile [time][ch] ------
  f16* g = &gl[wave * (16 * 64)];
#pragma unroll
  for (int m = 0; m < 4; ++m) {
    union { f16 h[8]; int4 v; } pk;
#pragma unroll
    for (int r = 0; r < 8; ++r)
      pk.h[r] = (f16)gated(acc[m][r]);
    *reinterpret_cast<int4*>(&g[n * 64 + m * 16 + hi * 8]) = pk.v;
  }
  __builtin_amdgcn_wave_barrier();   // ds stores precede ds loads (same wave)

  // ---- stage 2: res = Wres@g + bres ; sk = Wskip@g + bskip (K=64) --------
  v8f racc[4], sacc[4];
#pragma unroll
  for (int m = 0; m < 4; ++m) {
    const int cb = m * 16 + hi * 8;
#pragma unroll
    for (int r = 0; r < 8; ++r) {
      racc[m][r] = bres[cb + r];
      sacc[m][r] = bskip[cb + r];
    }
  }
#pragma unroll
  for (int kf = 0; kf < 2; ++kf) {
    const v16h gf = ld_fragB(&g[n * 64 + kf * 32], lane);
#pragma unroll
    for (int m = 0; m < 4; ++m) {
      const v16h ar = ld_fragA(&wr[(m * 16 + n) * 64 + kf * 32], lane);
      racc[m] = __builtin_amdgcn_wmma_f32_16x16x32_f16(
          false, ar, false, gf, (short)0, racc[m], false, false);
      const v16h as = ld_fragA(&wk[(m * 16 + n) * 64 + kf * 32], lane);
      sacc[m] = __builtin_amdgcn_wmma_f32_16x16x32_f16(
          false, as, false, gf, (short)0, sacc[m], false, false);
    }
  }

  // ---- residual add + store hOut -----------------------------------------
  const f16* rrow = hIn  + ((size_t)b * TPAD_ + (size_t)(t0 + n + d)) * 64;
  f16*       orow = hOut + ((size_t)b * TPAD_ + (size_t)(t0 + n)) * 64;
#pragma unroll
  for (int m = 0; m < 4; ++m) {
    const int cb = m * 16 + hi * 8;
    union { f16 h[8]; int4 v; } ri, ro;
    ri.v = *reinterpret_cast<const int4*>(rrow + cb);
#pragma unroll
    for (int r = 0; r < 8; ++r)
      ro.h[r] = (f16)(racc[m][r] + (float)ri.h[r]);
    *reinterpret_cast<int4*>(orow + cb) = ro.v;
  }

  // ---- skip accumulation (each (b,t) owned by exactly one wave: plain RMW)
  const int ts = t0 + n - skipOff;
  if (ts >= 0 && ts < SKIPN_) {
    float* sp = skip + ((size_t)b * SKIPN_ + ts) * 64;
#pragma unroll
    for (int m = 0; m < 4; ++m) {
      const int cb = m * 16 + hi * 8;
      float4 a0 = *reinterpret_cast<float4*>(sp + cb);
      float4 a1 = *reinterpret_cast<float4*>(sp + cb + 4);
      a0.x += sacc[m][0]; a0.y += sacc[m][1]; a0.z += sacc[m][2]; a0.w += sacc[m][3];
      a1.x += sacc[m][4]; a1.y += sacc[m][5]; a1.z += sacc[m][6]; a1.w += sacc[m][7];
      *reinterpret_cast<float4*>(sp + cb)     = a0;
      *reinterpret_cast<float4*>(sp + cb + 4) = a1;
    }
  }
}

// ---------------------------------------------------------------------------
// DenseNet head: relu -> W_d1 -> relu -> W_d2 -> softmax(ch); writes the
// channel-major flatten layout o_cm[b*65536 + c*1024 + ts] that FC0 consumes.
// ---------------------------------------------------------------------------
__global__ __launch_bounds__(256)
void k_dense(const float* __restrict__ skip, const float* __restrict__ Wd1,
             const float* __restrict__ bd1, const float* __restrict__ Wd2,
             const float* __restrict__ bd2, float* __restrict__ o_cm) {
  __shared__ float w1[64 * 64], w2[64 * 64], bb1[64], bb2[64];
  for (int i = threadIdx.x; i < 64 * 64; i += 256) { w1[i] = Wd1[i]; w2[i] = Wd2[i]; }
  if (threadIdx.x < 64) { bb1[threadIdx.x] = bd1[threadIdx.x]; bb2[threadIdx.x] = bd2[threadIdx.x]; }
  __syncthreads();

  const int idx = blockIdx.x * 256 + threadIdx.x;   // exactly B_*SKIPN_ threads
  const int b = idx >> 10, ts = idx & 1023;

  float s[64];
#pragma unroll
  for (int c = 0; c < 64; ++c) {
    const float v = skip[((size_t)b * SKIPN_ + ts) * 64 + c];
    s[c] = v > 0.0f ? v : 0.0f;
  }
  float o1[64];
#pragma unroll 4
  for (int o = 0; o < 64; ++o) {
    float a = bb1[o];
#pragma unroll
    for (int c = 0; c < 64; ++c) a += w1[o * 64 + c] * s[c];
    o1[o] = a > 0.0f ? a : 0.0f;
  }
  float o2[64];
  float mx = -1e30f;
#pragma unroll 4
  for (int o = 0; o < 64; ++o) {
    float a = bb2[o];
#pragma unroll
    for (int c = 0; c < 64; ++c) a += w2[o * 64 + c] * o1[c];
    o2[o] = a;
    mx = fmaxf(mx, a);
  }
  float sum = 0.0f;
#pragma unroll
  for (int o = 0; o < 64; ++o) { o2[o] = __expf(o2[o] - mx); sum += o2[o]; }
  const float inv = 1.0f / sum;
  float* dst = o_cm + (size_t)b * (SKIPN_ * 64) + ts;
#pragma unroll
  for (int c = 0; c < 64; ++c) dst[c * SKIPN_] = o2[c] * inv;
}

// ---------------------------------------------------------------------------
// FC partial GEMV: out[b][j] += sum_k in[b][k]*W[j][k] over a K chunk.
// grid (M, B, ksplit), block 64. HBM-bound on W (FC0: 134 MB) -> K-split.
// ---------------------------------------------------------------------------
__global__ __launch_bounds__(64)
void k_fc(const float* __restrict__ in, const float* __restrict__ W,
          float* __restrict__ outAcc, int M, int K, int Kstride, int ksplit) {
  __shared__ float red[64];
  const int j = blockIdx.x, b = blockIdx.y;
  const int chunk = K / ksplit;
  const int k0 = blockIdx.z * chunk;
  const float* ip = in + (size_t)b * Kstride + k0;
  const float* wp = W + (size_t)j * K + k0;
  float p = 0.0f;
  for (int k = threadIdx.x; k < chunk; k += 64) p += ip[k] * wp[k];
  red[threadIdx.x] = p;
  __syncthreads();
  if (threadIdx.x == 0) {
    float sum = 0.0f;
#pragma unroll
    for (int i = 0; i < 64; ++i) sum += red[i];
    atomicAdd(&outAcc[(size_t)b * M + j], sum);
  }
}

__global__ void k_biasact(const float* __restrict__ acc, const float* __restrict__ bias,
                          float* __restrict__ dst, int M, int total, int doTanh) {
  const int i = blockIdx.x * 256 + threadIdx.x;
  if (i >= total) return;
  const float v = acc[i] + bias[i % M];
  dst[i] = doTanh ? tanhf(v) : v;
}

// ---------------------------------------------------------------------------
extern "C" void kernel_launch(void* const* d_in, const int* in_sizes, int n_in,
                              void* d_out, int out_size, void* d_ws, size_t ws_size,
                              hipStream_t stream) {
  (void)in_sizes; (void)n_in; (void)out_size; (void)ws_size;
  const float* x     = (const float*)d_in[0];
  const float* Wc    = (const float*)d_in[1];
  const float* Wdil  = (const float*)d_in[2];
  const float* Wres  = (const float*)d_in[3];
  const float* bres  = (const float*)d_in[4];
  const float* Wskip = (const float*)d_in[5];
  const float* bskip = (const float*)d_in[6];
  const float* Wd1   = (const float*)d_in[7];
  const float* bd1   = (const float*)d_in[8];
  const float* Wd2   = (const float*)d_in[9];
  const float* bd2   = (const float*)d_in[10];
  const float* Wfc0  = (const float*)d_in[11];
  const float* bfc0  = (const float*)d_in[12];
  const float* Wfc1  = (const float*)d_in[13];
  const float* bfc1  = (const float*)d_in[14];
  const float* Wfc2  = (const float*)d_in[15];
  const float* bfc2  = (const float*)d_in[16];
  float* out = (float*)d_out;

  char* ws = (char*)d_ws;
  size_t off = 0;
  auto take = [&](size_t bytes) {
    char* p = ws + off;
    off = (off + bytes + 255) & ~(size_t)255;
    return p;
  };
  const size_t hbytes = (size_t)B_ * TPAD_ * 64 * sizeof(f16);
  f16*   hA   = (f16*)take(hbytes);
  f16*   hB   = (f16*)take(hbytes);
  f16*   wAll = (f16*)take((size_t)NBLK_ * WBLKH_ * sizeof(f16));
  float* skip = (float*)take((size_t)B_ * SKIPN_ * 64 * sizeof(float));
  float* o_cm = (float*)take((size_t)B_ * SKIPN_ * 64 * sizeof(float));
  float* z0a  = (float*)take((size_t)B_ * DIMFC_ * sizeof(float));
  float* z0   = (float*)take((size_t)B_ * DIMFC_ * sizeof(float));
  float* z1a  = (float*)take((size_t)B_ * DIMFC_ * sizeof(float));
  float* z1   = (float*)take((size_t)B_ * DIMFC_ * sizeof(float));
  float* z2a  = (float*)take((size_t)B_ * OUTN_ * sizeof(float));

  hipMemsetAsync(skip, 0, (size_t)B_ * SKIPN_ * 64 * sizeof(float), stream);
  hipMemsetAsync(z0a, 0, (size_t)B_ * DIMFC_ * sizeof(float), stream);
  hipMemsetAsync(z1a, 0, (size_t)B_ * DIMFC_ * sizeof(float), stream);
  hipMemsetAsync(z2a, 0, (size_t)B_ * OUTN_ * sizeof(float), stream);

  {
    const int total = NBLK_ * WBLKH_;
    k_wcvt<<<(total + 255) / 256, 256, 0, stream>>>(Wdil, Wres, Wskip, wAll);
  }
  {
    const int total = B_ * T0_ * CR_;
    k_causal<<<(total + 255) / 256, 256, 0, stream>>>(x, Wc, hA);
  }

  f16* cur = hA; f16* nxt = hB;
  int Tin = T0_;
  for (int i = 0; i < NBLK_; ++i) {
    const int d = 1 << (i % 10);
    const int Tout = Tin - d;
    dim3 grid((Tout + 127) / 128, B_);
    k_block<<<grid, 256, 0, stream>>>(
        cur, nxt, skip, wAll + (size_t)i * WBLKH_,
        bres + (size_t)i * 64, bskip + (size_t)i * 64,
        d, Tout, Tout - SKIPN_);
    f16* t = cur; cur = nxt; nxt = t;
    Tin = Tout;
  }

  k_dense<<<(B_ * SKIPN_) / 256, 256, 0, stream>>>(skip, Wd1, bd1, Wd2, bd2, o_cm);

  {
    dim3 g(DIMFC_, B_, 8);
    k_fc<<<g, 64, 0, stream>>>(o_cm, Wfc0, z0a, DIMFC_, SKIPN_ * 64, SKIPN_ * 64, 8);
  }
  k_biasact<<<(B_ * DIMFC_ + 255) / 256, 256, 0, stream>>>(z0a, bfc0, z0, DIMFC_, B_ * DIMFC_, 1);
  {
    dim3 g(DIMFC_, B_, 1);
    k_fc<<<g, 64, 0, stream>>>(z0, Wfc1, z1a, DIMFC_, DIMFC_, DIMFC_, 1);
  }
  k_biasact<<<(B_ * DIMFC_ + 255) / 256, 256, 0, stream>>>(z1a, bfc1, z1, DIMFC_, B_ * DIMFC_, 1);
  {
    dim3 g(OUTN_, B_, 1);
    k_fc<<<g, 64, 0, stream>>>(z1, Wfc2, z2a, OUTN_, DIMFC_, DIMFC_, 1);
  }
  k_biasact<<<(B_ * OUTN_ + 255) / 256, 256, 0, stream>>>(z2a, bfc2, out, OUTN_, B_ * OUTN_, 0);
}
